// KNRM_57483842290258
// MI455X (gfx1250) — matrix operand order repaired and checked by reading, not compiled, not run on hardware
//
#include <hip/hip_runtime.h>
#include <hip/hip_bf16.h>
#include <math.h>

typedef _Float16 v16h __attribute__((ext_vector_type(16)));
typedef _Float16 h2   __attribute__((ext_vector_type(2)));
typedef float    v8f  __attribute__((ext_vector_type(8)));

#define LQ 20
#define LD 200
#define EDIM 128
#define NK 21
#define QR 32     // padded query rows (2 row tiles of 16)
#define DR 208    // padded doc rows (13 col tiles of 16)
#define ST 130    // f16 row stride in LDS (= 65 dwords, bank-conflict-free)

// 16-lane sum within each half of the wave, done purely in the VALU via DPP16.
// Stage order: xor1 (quad_perm 1,0,3,2), xor2 (quad_perm 2,3,0,1), then
// ROW_HALF_MIRROR (pairs quads after quads are uniform) and ROW_MIRROR
// (pairs octets after octets are uniform). No DS ops, no dscnt waits.
template <int CTRL>
static __device__ __forceinline__ float dpp_add(float x) {
  int y = __builtin_amdgcn_update_dpp(0, __float_as_int(x), CTRL, 0xF, 0xF, true);
  return x + __int_as_float(y);
}
static __device__ __forceinline__ float row16_sum(float x) {
  x = dpp_add<0xB1>(x);    // quad_perm [1,0,3,2]  : xor 1
  x = dpp_add<0x4E>(x);    // quad_perm [2,3,0,1]  : xor 2
  x = dpp_add<0x141>(x);   // ROW_HALF_MIRROR      : xor 4 (post-uniform quads)
  x = dpp_add<0x140>(x);   // ROW_MIRROR           : xor 8 (post-uniform octets)
  return x;
}

// ---------------------------------------------------------------------------
// Phase 1: per (pair, batch) block: gather -> f16 LDS, WMMA cosine matrix,
// RBF kernel pooling with DPP butterfly reduction, log1p sum -> [B,21]
// ---------------------------------------------------------------------------
__global__ __launch_bounds__(256) void knrm_phase1(
    const float* __restrict__ emb,
    const int*   __restrict__ qids,
    const int*   __restrict__ dids,
    float*       __restrict__ outk)   // [B, NK]
{
  __shared__ _Float16 sQ[QR * ST];
  __shared__ _Float16 sD[DR * ST];
  __shared__ float sInvQ[QR];
  __shared__ float sInvD[DR];
  __shared__ float sS[LQ][NK];

  const int b   = blockIdx.x;
  const int tid = threadIdx.x;

  for (int i = tid; i < LQ * NK; i += 256) (&sS[0][0])[i] = 0.0f;

  // ---- gather Q rows -> f16 LDS (rows >= LQ zero-filled) ----
  for (int i = tid; i < QR * 32; i += 256) {
    const int row = i >> 5, c4 = i & 31;
    float4 v = make_float4(0.f, 0.f, 0.f, 0.f);
    if (row < LQ) {
      const int id = qids[b * LQ + row];
      v = ((const float4*)(emb + (size_t)id * EDIM))[c4];
    }
    h2* dst = (h2*)&sQ[row * ST + c4 * 4];
    h2 p0, p1;
    p0[0] = (_Float16)v.x; p0[1] = (_Float16)v.y;
    p1[0] = (_Float16)v.z; p1[1] = (_Float16)v.w;
    dst[0] = p0; dst[1] = p1;
  }
  // ---- gather D rows ----
  for (int i = tid; i < DR * 32; i += 256) {
    const int row = i >> 5, c4 = i & 31;
    float4 v = make_float4(0.f, 0.f, 0.f, 0.f);
    if (row < LD) {
      const int id = dids[b * LD + row];
      v = ((const float4*)(emb + (size_t)id * EDIM))[c4];
    }
    h2* dst = (h2*)&sD[row * ST + c4 * 4];
    h2 p0, p1;
    p0[0] = (_Float16)v.x; p0[1] = (_Float16)v.y;
    p1[0] = (_Float16)v.z; p1[1] = (_Float16)v.w;
    dst[0] = p0; dst[1] = p1;
  }
  __syncthreads();

  // ---- inverse norms computed from the same f16 data the WMMA consumes ----
  for (int r = tid; r < DR + QR; r += 256) {
    const h2* src = (r < DR) ? (const h2*)&sD[r * ST]
                             : (const h2*)&sQ[(r - DR) * ST];
    float s = 0.f;
    #pragma unroll 8
    for (int i = 0; i < EDIM / 2; ++i) {
      h2 x = src[i];
      float x0 = (float)x[0], x1 = (float)x[1];
      s += x0 * x0 + x1 * x1;
    }
    const float inv = rsqrtf(s + 1e-6f);
    if (r < DR) sInvD[r] = inv; else sInvQ[r - DR] = inv;
  }
  __syncthreads();

  const int wave = tid >> 5;
  const int lane = tid & 31;
  const int rt   = wave & 1;    // row tile (query rows rt*16 ..)
  const int cset = wave >> 1;   // 0..3 -> col tiles cset, cset+4, cset+8, ...
  const int lm   = lane & 15;
  const int hi   = lane >> 4;

  // A fragments (16x32 f16 layout): lane<16 -> K {0..7, 16..23} (+K0),
  // lane>=16 -> K {8..15, 24..31}. Hoisted over the ct loop.
  v16h afrag[4];
  {
    const _Float16* qrow = &sQ[(rt * 16 + lm) * ST];
    #pragma unroll
    for (int ks = 0; ks < 4; ++ks) {
      const int K0 = ks * 32;
      const h2* p0 = (const h2*)(qrow + K0 + 8 * hi);
      const h2* p1 = (const h2*)(qrow + K0 + 16 + 8 * hi);
      v16h a;
      #pragma unroll
      for (int e = 0; e < 4; ++e) {
        h2 x = p0[e]; a[2*e]     = x[0]; a[2*e + 1]     = x[1];
        h2 y = p1[e]; a[8 + 2*e] = y[0]; a[8 + 2*e + 1] = y[1];
      }
      afrag[ks] = a;
    }
  }

  // Inverse query norms for this lane's 8 output rows (hoisted).
  float qinv[8];
  #pragma unroll
  for (int j = 0; j < 8; ++j) qinv[j] = sInvQ[rt * 16 + j + 8 * hi];

  // Per-lane accumulators: after the butterfly every lane of a half holds the
  // full 16-column sum, so lane lm keeps kernels k==lm and k==16+lm.
  float accLo[8], accHi[8];
  #pragma unroll
  for (int j = 0; j < 8; ++j) { accLo[j] = 0.f; accHi[j] = 0.f; }

  for (int ct = cset; ct < 13; ct += 4) {
    const int n = ct * 16 + lm;
    const _Float16* drow = &sD[n * ST];

    v8f c = {0.f, 0.f, 0.f, 0.f, 0.f, 0.f, 0.f, 0.f};
    #pragma unroll
    for (int ks = 0; ks < 4; ++ks) {
      // B fragment (32x16 f16): lane<16 -> K {0..7, 16..23} (+K0), lane>=16 -> K {8..15, 24..31}
      const h2* p = (const h2*)(drow + ks * 32 + 8 * hi);
      v16h bf;
      #pragma unroll
      for (int e = 0; e < 8; ++e) { h2 x = p[(e & 3) + 8 * (e >> 2)]; bf[2*e] = x[0]; bf[2*e+1] = x[1]; }
      c = __builtin_amdgcn_wmma_f32_16x16x32_f16(false, afrag[ks], false, bf,
                                                 (short)0, c, false, false);
    }

    const float invd   = sInvD[n];
    const bool  nvalid = (n < LD);

    // Cosine values; poison padded doc columns so every kernel underflows to 0.
    float Mv[8];
    #pragma unroll
    for (int j = 0; j < 8; ++j) {
      const float m = c[j] * (invd * qinv[j]);
      Mv[j] = nvalid ? m : 1.0e4f;
    }

    #pragma unroll
    for (int k = 0; k < NK; ++k) {
      // phi = exp(-(M-mu)^2 / (2 sig^2)) = exp2((M-mu)^2 * ck)
      const float mu = (k < 20) ? (-0.95f + 0.1f * (float)k) : 1.0f;
      const float ck = (k < 20) ? -72.134750f : -721347.52f;  // -log2(e)/(2s^2)
      const bool sel = (lm == ((k < 16) ? k : (k - 16)));     // one v_cmp per k
      #pragma unroll
      for (int j = 0; j < 8; ++j) {
        const float t = Mv[j] - mu;
        float phi = __builtin_amdgcn_exp2f(t * t * ck);       // raw v_exp_f32
        phi = row16_sum(phi);                                 // DPP butterfly
        if (k < 16) accLo[j] += sel ? phi : 0.f;
        else        accHi[j] += sel ? phi : 0.f;
      }
    }
  }

  // Flush per-lane accumulators: all 32 lanes hit distinct (m,k) -> no
  // address conflicts; 16 ds_add_f32 per wave total.
  #pragma unroll
  for (int j = 0; j < 8; ++j) {
    const int m = rt * 16 + j + 8 * hi;
    if (m < LQ) {
      atomicAdd(&sS[m][lm], accLo[j]);
      if (lm < 5) atomicAdd(&sS[m][16 + lm], accHi[j]);
    }
  }
  __syncthreads();

  if (tid < NK) {
    float acc = 0.f;
    #pragma unroll
    for (int q = 0; q < LQ; ++q) acc += log1pf(sS[q][tid]);
    outk[b * NK + tid] = acc;
  }
}

// ---------------------------------------------------------------------------
// Phase 2: MLP on both pair features + sigmoid(l1 - l2); one thread per batch
// ---------------------------------------------------------------------------
__global__ __launch_bounds__(256) void knrm_phase2(
    const float* __restrict__ k1, const float* __restrict__ k2,
    const float* __restrict__ W1, const float* __restrict__ b1,
    const float* __restrict__ W2, const float* __restrict__ b2,
    const float* __restrict__ W3, const float* __restrict__ b3,
    float* __restrict__ out, int B)
{
  const int b = blockIdx.x * blockDim.x + threadIdx.x;
  if (b >= B) return;

  float l[2];
  #pragma unroll
  for (int p = 0; p < 2; ++p) {
    const float* x = (p ? k2 : k1) + b * NK;
    float h1[10];
    #pragma unroll
    for (int i = 0; i < 10; ++i) {
      float acc = b1[i];
      for (int j = 0; j < NK; ++j) {
        float xv = x[j]; xv = xv > 0.f ? xv : 0.f;       // relu on input
        acc += xv * W1[i * NK + j];
      }
      h1[i] = acc > 0.f ? acc : 0.f;
    }
    float hh[5];
    #pragma unroll
    for (int i = 0; i < 5; ++i) {
      float acc = b2[i];
      for (int j = 0; j < 10; ++j) acc += h1[j] * W2[i * 10 + j];
      hh[i] = acc > 0.f ? acc : 0.f;
    }
    float acc = b3[0];
    #pragma unroll
    for (int j = 0; j < 5; ++j) acc += hh[j] * W3[j];
    l[p] = acc;
  }
  out[b] = 1.0f / (1.0f + expf(-(l[0] - l[1])));
}

extern "C" void kernel_launch(void* const* d_in, const int* in_sizes, int n_in,
                              void* d_out, int out_size, void* d_ws, size_t ws_size,
                              hipStream_t stream) {
  const int*   q1  = (const int*)d_in[0];
  const int*   d1  = (const int*)d_in[1];
  const int*   q2  = (const int*)d_in[2];
  const int*   d2  = (const int*)d_in[3];
  const float* emb = (const float*)d_in[4];
  const float* W1  = (const float*)d_in[5];
  const float* b1  = (const float*)d_in[6];
  const float* W2  = (const float*)d_in[7];
  const float* b2  = (const float*)d_in[8];
  const float* W3  = (const float*)d_in[9];
  const float* b3  = (const float*)d_in[10];

  const int B = in_sizes[0] / LQ;   // 1024

  float* ws = (float*)d_ws;
  float* k1 = ws;                    // [B, NK]
  float* k2 = ws + (size_t)B * NK;   // [B, NK]

  knrm_phase1<<<B, 256, 0, stream>>>(emb, q1, d1, k1);
  knrm_phase1<<<B, 256, 0, stream>>>(emb, q2, d2, k2);
  knrm_phase2<<<(B + 255) / 256, 256, 0, stream>>>(
      k1, k2, W1, b1, W2, b2, W3, b3, (float*)d_out, B);
}
